// ModuleRelProp1_24790551232700
// MI455X (gfx1250) — matrix operand order repaired and checked by reading, not compiled, not run on hardware
//
#include <hip/hip_runtime.h>

#define BB 2
#define NN 2048
#define KK 256
#define DD 768
#define HH 150
#define LL 11
#define NBK 10
#define HP 160   // H padded to multiple of 32 (WMMA K-step)
#define LP 16    // L padded to one WMMA N-tile

typedef __attribute__((ext_vector_type(16))) _Float16 v16h;
typedef __attribute__((ext_vector_type(8)))  _Float16 v8h;
typedef __attribute__((ext_vector_type(8)))  float    v8f;

// ---------------------------------------------------------------------------
// WMMA fragment loaders (CDNA5 16-bit A 16x32 / B 32x16 VGPR striping).
// A: lane&15 = row M; (lane>>4)*8 = K sub-offset; VGPRs 0-3 hold K..K+7,
//    VGPRs 4-7 hold K+16..K+23  -> two contiguous 16B LDS loads.
// B: stored transposed (wT[n][k]) so lane&15 = col N, (lane>>4)*8 = K sub-offset,
//    K..K+7 and K+16..K+23 (same as A) -> two contiguous 16B LDS loads.
// ---------------------------------------------------------------------------
static __device__ __forceinline__ v16h load_frag_a(const _Float16* base, int mrow,
                                                   int kbase, int lane) {
  const _Float16* rp = base + (size_t)(mrow + (lane & 15)) * HP + kbase + ((lane >> 4) << 3);
  v8h lo = *(const v8h*)rp;
  v8h hi = *(const v8h*)(rp + 16);
  v16h a;
#pragma unroll
  for (int i = 0; i < 8; ++i) { a[i] = lo[i]; a[i + 8] = hi[i]; }
  return a;
}

static __device__ __forceinline__ v16h load_frag_b(const _Float16* baseT, int ncol,
                                                   int kbase, int lane) {
  const _Float16* rp = baseT + (size_t)(ncol + (lane & 15)) * HP + kbase + ((lane >> 4) << 3);
  v8h lo = *(const v8h*)rp;
  v8h hi = *(const v8h*)(rp + 16);
  v16h b;
#pragma unroll
  for (int i = 0; i < 8; ++i) { b[i] = lo[i]; b[i + 8] = hi[i]; }
  return b;
}

// Async global -> LDS 128-bit copy (CDNA5, ASYNCcnt-tracked, no VGPR data).
static __device__ __forceinline__ void async_copy_b128(unsigned lds_addr,
                                                       unsigned long long gaddr) {
  asm volatile("global_load_async_to_lds_b128 %0, %1, off"
               :: "v"(lds_addr), "v"(gaddr) : "memory");
}

// ---------------------------------------------------------------------------
// One-time weight prep: w_h [H][H] -> f16 transposed padded [HP][HP],
//                       w_out [H][L] -> f16 transposed padded [LP][HP].
// Zero padding makes the K=160 WMMA accumulation exact.
// ---------------------------------------------------------------------------
__global__ void prep_weights_kernel(const float* __restrict__ wh,
                                    const float* __restrict__ wo,
                                    _Float16* __restrict__ whT,
                                    _Float16* __restrict__ woT) {
  const int idx = blockIdx.x * 256 + threadIdx.x;
  if (idx < HP * HP) {
    const int c = idx / HP, k = idx % HP;
    const float v = (c < HH && k < HH) ? wh[k * HH + c] : 0.f;
    whT[idx] = (_Float16)v;
  }
  if (idx < LP * HP) {
    const int l = idx / HP, k = idx % HP;
    const float v = (l < LL && k < HH) ? wo[k * LL + l] : 0.f;
    woT[idx] = (_Float16)v;
  }
}

// ---------------------------------------------------------------------------
// left/right/pruner projections: one block per (b,k) row.
// left/right are written zero-padded with stride HP so the pairwise kernel
// can use unguarded float4 loads.
// ---------------------------------------------------------------------------
__global__ __launch_bounds__(256) void proj_kernel(
    const float* __restrict__ u,
    const float* __restrict__ wl, const float* __restrict__ bl,
    const float* __restrict__ wr, const float* __restrict__ br,
    const float* __restrict__ prw, const float* __restrict__ prb,
    float* __restrict__ left, float* __restrict__ right, float* __restrict__ ss) {
  __shared__ float urow[DD];
  __shared__ float red[256];
  const int row = blockIdx.x;  // b*K + k
  const float* up = u + (size_t)row * DD;
  for (int d = threadIdx.x; d < DD; d += 256) urow[d] = up[d];
  __syncthreads();
  const int h = threadIdx.x;
  if (h < HP) {
    if (h < HH) {
      float al = bl[h], ar = br[h];
      for (int d = 0; d < DD; ++d) {
        const float uv = urow[d];
        al = fmaf(uv, wl[d * HH + h], al);
        ar = fmaf(uv, wr[d * HH + h], ar);
      }
      left[(size_t)row * HP + h]  = al;
      right[(size_t)row * HP + h] = ar;
    } else {
      left[(size_t)row * HP + h]  = 0.f;
      right[(size_t)row * HP + h] = 0.f;
    }
  }
  float p = 0.f;
  for (int d = threadIdx.x; d < DD; d += 256) p = fmaf(urow[d], prw[d], p);
  red[threadIdx.x] = p;
  __syncthreads();
  for (int s = 128; s > 0; s >>= 1) {
    if (threadIdx.x < s) red[threadIdx.x] += red[threadIdx.x + s];
    __syncthreads();
  }
  if (threadIdx.x == 0) ss[row] = red[0] + prb[0];
}

// ---------------------------------------------------------------------------
// Fused pairwise FF: block = 16x16 (i,j) pair tile (256 GEMM rows).
//   h1 = relu(left_i + right_j + demb[bin])  -> LDS f16 [256][HP]
//   h2 = relu(h1 @ w_h + b_h)                -> per-wave LDS f16 [32][HP]
//   rel = h2 @ w_out + b_out + ss_i + ss_j   -> global
// Weights arrive via global_load_async_to_lds_b128 overlapped with the h1
// build; each wave owns two M-tiles so every B fragment feeds 2 WMMAs.
// ---------------------------------------------------------------------------
__global__ __launch_bounds__(256) void pair_ff_kernel(
    const float* __restrict__ left, const float* __restrict__ right,
    const float* __restrict__ ss, const int* __restrict__ span_begin,
    const float* __restrict__ dist_emb, const float* __restrict__ b_h,
    const float* __restrict__ b_out,
    const _Float16* __restrict__ whT_g, const _Float16* __restrict__ woT_g,
    float* __restrict__ relout) {
  extern __shared__ char smem[];
  _Float16* h1    = (_Float16*)smem;          // [256][HP]      80 KB
  _Float16* whT_s = h1 + 256 * HP;            // [HP][HP]       51 KB (transposed)
  _Float16* woT_s = whT_s + HP * HP;          // [LP][HP]        5 KB (transposed)
  _Float16* h2all = woT_s + LP * HP;          // 8 x [32][HP]   80 KB
  float*    demb_s = (float*)(h2all + 8 * 32 * HP);  // [NBK][HP]  6.4 KB

  const int tid = threadIdx.x;
  const int b   = blockIdx.z;
  const int i0  = blockIdx.x * 16;
  const int j0  = blockIdx.y * 16;

  // 1) Kick off async DMA of the f16 weights into LDS (ASYNCcnt-tracked).
  {
    const unsigned wh_lds = (unsigned)(uintptr_t)(void*)whT_s;
    const unsigned long long wh_g = (unsigned long long)whT_g;
    for (int idx = tid; idx < (HP * HP * 2) / 16; idx += 256)
      async_copy_b128(wh_lds + idx * 16, wh_g + idx * 16);
    const unsigned wo_lds = (unsigned)(uintptr_t)(void*)woT_s;
    const unsigned long long wo_g = (unsigned long long)woT_g;
    for (int idx = tid; idx < (LP * HP * 2) / 16; idx += 256)
      async_copy_b128(wo_lds + idx * 16, wo_g + idx * 16);
  }

  // 2) Stage dist_emb zero-padded as f32.
  for (int idx = tid; idx < NBK * HP; idx += 256) {
    const int bn = idx / HP, c = idx - bn * HP;
    demb_s[idx] = (c < HH) ? dist_emb[bn * HH + c] : 0.f;
  }
  __syncthreads();  // demb visible

  // 3) Build h1 tile (overlaps with in-flight weight DMA): one pair per thread.
  {
    const int p  = tid;
    const int iL = p >> 4, jL = p & 15;
    const int gi = i0 + iL, gj = j0 + jL;
    int diff = span_begin[b * KK + gi] - span_begin[b * KK + gj];
    diff = diff < 0 ? -diff : diff;
    const int bin = diff < (NBK - 1) ? diff : (NBK - 1);
    const float* lp = left + ((size_t)b * KK + gi) * HP;
    const float* rp = right + ((size_t)b * KK + gj) * HP;
    const float* dp = demb_s + (size_t)bin * HP;
    _Float16* hr = h1 + (size_t)p * HP;
#pragma unroll 4
    for (int c = 0; c < HP; c += 8) {
      const float4 l0 = *(const float4*)(lp + c);
      const float4 l1 = *(const float4*)(lp + c + 4);
      const float4 r0 = *(const float4*)(rp + c);
      const float4 r1 = *(const float4*)(rp + c + 4);
      const float4 e0 = *(const float4*)(dp + c);
      const float4 e1 = *(const float4*)(dp + c + 4);
      v8h o;
      o[0] = (_Float16)fmaxf(l0.x + r0.x + e0.x, 0.f);
      o[1] = (_Float16)fmaxf(l0.y + r0.y + e0.y, 0.f);
      o[2] = (_Float16)fmaxf(l0.z + r0.z + e0.z, 0.f);
      o[3] = (_Float16)fmaxf(l0.w + r0.w + e0.w, 0.f);
      o[4] = (_Float16)fmaxf(l1.x + r1.x + e1.x, 0.f);
      o[5] = (_Float16)fmaxf(l1.y + r1.y + e1.y, 0.f);
      o[6] = (_Float16)fmaxf(l1.z + r1.z + e1.z, 0.f);
      o[7] = (_Float16)fmaxf(l1.w + r1.w + e1.w, 0.f);
      *(v8h*)(hr + c) = o;
    }
  }

  // 4) Drain the async weight DMA, then make everything block-visible.
  asm volatile("s_wait_asynccnt 0" ::: "memory");
  __syncthreads();

  const int wave = tid >> 5;
  const int lane = tid & 31;
  const int nlan = lane & 15;
  _Float16* h2 = h2all + (size_t)wave * 32 * HP;
  const int mrow0 = wave * 32;      // wave owns M-tiles 2*wave, 2*wave+1

  // ---- GEMM1: [32 x HP] x [HP x HP] ----
  v8f acc[2][10];
#pragma unroll
  for (int nt = 0; nt < 10; ++nt) {
    const int hcol = nt * 16 + nlan;
    const float bh = (hcol < HH) ? b_h[hcol] : 0.f;
#pragma unroll
    for (int q = 0; q < 8; ++q) { acc[0][nt][q] = bh; acc[1][nt][q] = bh; }
  }
#pragma unroll
  for (int kt = 0; kt < 5; ++kt) {
    const v16h a0 = load_frag_a(h1, mrow0,      kt * 32, lane);
    const v16h a1 = load_frag_a(h1, mrow0 + 16, kt * 32, lane);
#pragma unroll
    for (int nt = 0; nt < 10; ++nt) {
      const v16h bf = load_frag_b(whT_s, nt * 16, kt * 32, lane);
      acc[0][nt] = __builtin_amdgcn_wmma_f32_16x16x32_f16(
          false, a0, false, bf, (short)0, acc[0][nt], false, false);
      acc[1][nt] = __builtin_amdgcn_wmma_f32_16x16x32_f16(
          false, a1, false, bf, (short)0, acc[1][nt], false, false);
    }
  }
  // relu -> f16 -> wave-private LDS tile (C/D layout: VGPR v -> row v or v+8)
#pragma unroll
  for (int mi = 0; mi < 2; ++mi) {
#pragma unroll
    for (int nt = 0; nt < 10; ++nt) {
      const int n = nt * 16 + nlan;
#pragma unroll
      for (int v = 0; v < 8; ++v) {
        const int m = mi * 16 + ((lane < 16) ? v : v + 8);
        const float x = acc[mi][nt][v];
        h2[m * HP + n] = (_Float16)(x > 0.f ? x : 0.f);
      }
    }
  }
  asm volatile("s_wait_dscnt 0" ::: "memory");  // h2 store -> load ordering

  // ---- GEMM2: [32 x HP] x [HP x LP] ----
  v8f acc2[2];
  {
    const float bo = (nlan < LL) ? b_out[nlan] : 0.f;
#pragma unroll
    for (int q = 0; q < 8; ++q) { acc2[0][q] = bo; acc2[1][q] = bo; }
  }
#pragma unroll
  for (int kt = 0; kt < 5; ++kt) {
    const v16h bf = load_frag_b(woT_s, 0, kt * 32, lane);
    const v16h a0 = load_frag_a(h2, 0,  kt * 32, lane);
    const v16h a1 = load_frag_a(h2, 16, kt * 32, lane);
    acc2[0] = __builtin_amdgcn_wmma_f32_16x16x32_f16(
        false, a0, false, bf, (short)0, acc2[0], false, false);
    acc2[1] = __builtin_amdgcn_wmma_f32_16x16x32_f16(
        false, a1, false, bf, (short)0, acc2[1], false, false);
  }

  // add unary pruner scores and store rel[b, i, j, l]
#pragma unroll
  for (int mi = 0; mi < 2; ++mi) {
    const int gi  = i0 + wave * 2 + mi;
    const float ssi = ss[b * KK + gi];
    if (nlan < LL) {
#pragma unroll
      for (int v = 0; v < 8; ++v) {
        const int m  = (lane < 16) ? v : v + 8;
        const int gj = j0 + m;
        const float ssj = ss[b * KK + gj];
        relout[(((size_t)b * KK + gi) * KK + gj) * LL + nlan] = acc2[mi][v] + ssi + ssj;
      }
    }
  }
}

// ---------------------------------------------------------------------------
// Relation propagation + gate, restructured to avoid the [B,K,K,D] msg tensor:
//   ctxt[b,j,d] = (1/len) * sum_l a_w[l,d] * sum_i probs[b,i,j,l]*u[b,i,d]
// One block per (b,j); thread owns 3 consecutive d columns.
// ---------------------------------------------------------------------------
__global__ __launch_bounds__(256) void relprop_gate_kernel(
    const float* __restrict__ rel, const float* __restrict__ mask,
    const float* __restrict__ ucur, const float* __restrict__ a_w,
    const float* __restrict__ gate_w, const float* __restrict__ gate_b,
    const float* __restrict__ span_len, float* __restrict__ unext) {
  __shared__ float ptile[32][12];
  __shared__ float crow[DD];
  __shared__ float urow[DD];
  const int j = blockIdx.x, b = blockIdx.y;
  const int tid = threadIdx.x;
  const int d0 = tid * 3;

  float T[LL][3];
#pragma unroll
  for (int l = 0; l < LL; ++l) { T[l][0] = 0.f; T[l][1] = 0.f; T[l][2] = 0.f; }

  for (int ic = 0; ic < KK; ic += 32) {
    for (int idx = tid; idx < 32 * LL; idx += 256) {
      const int ii = idx / LL, l = idx - ii * LL;
      const size_t ri = ((size_t)b * KK + (ic + ii)) * KK + j;
      float r = rel[ri * LL + l];
      r = r > 0.f ? r : 0.f;
      ptile[ii][l] = r * mask[ri];
    }
    __syncthreads();
    for (int ii = 0; ii < 32; ++ii) {
      const float* up = ucur + ((size_t)b * KK + ic + ii) * DD + d0;
      const float u0 = up[0], u1 = up[1], u2 = up[2];
#pragma unroll
      for (int l = 0; l < LL; ++l) {
        const float p = ptile[ii][l];
        T[l][0] = fmaf(p, u0, T[l][0]);
        T[l][1] = fmaf(p, u1, T[l][1]);
        T[l][2] = fmaf(p, u2, T[l][2]);
      }
    }
    __syncthreads();
  }

  const float inv = 1.f / span_len[b];
  float c0 = 0.f, c1 = 0.f, c2 = 0.f;
#pragma unroll
  for (int l = 0; l < LL; ++l) {
    const float* aw = a_w + (size_t)l * DD + d0;
    c0 = fmaf(aw[0], T[l][0], c0);
    c1 = fmaf(aw[1], T[l][1], c1);
    c2 = fmaf(aw[2], T[l][2], c2);
  }
  c0 *= inv; c1 *= inv; c2 *= inv;
  crow[d0] = c0; crow[d0 + 1] = c1; crow[d0 + 2] = c2;
  for (int d = tid; d < DD; d += 256) urow[d] = ucur[((size_t)b * KK + j) * DD + d];
  __syncthreads();

  // gate GEMV: sigmoid([u, ctxt] @ gate_w + gate_b)
  float s0 = gate_b[d0], s1 = gate_b[d0 + 1], s2 = gate_b[d0 + 2];
  for (int c = 0; c < DD; ++c) {
    const float uv = urow[c];
    const float* gw = gate_w + (size_t)c * DD + d0;
    s0 = fmaf(uv, gw[0], s0); s1 = fmaf(uv, gw[1], s1); s2 = fmaf(uv, gw[2], s2);
  }
  for (int c = 0; c < DD; ++c) {
    const float cv = crow[c];
    const float* gw = gate_w + (size_t)(DD + c) * DD + d0;
    s0 = fmaf(cv, gw[0], s0); s1 = fmaf(cv, gw[1], s1); s2 = fmaf(cv, gw[2], s2);
  }
  const float g0 = 1.f / (1.f + __expf(-s0));
  const float g1 = 1.f / (1.f + __expf(-s1));
  const float g2 = 1.f / (1.f + __expf(-s2));
  float* op = unext + ((size_t)b * KK + j) * DD + d0;
  op[0] = g0 * urow[d0]     + (1.f - g0) * c0;
  op[1] = g1 * urow[d0 + 1] + (1.f - g1) * c1;
  op[2] = g2 * urow[d0 + 2] + (1.f - g2) * c2;
}

// ---------------------------------------------------------------------------
// Output assembly
// ---------------------------------------------------------------------------
__global__ void copy4_kernel(const float4* __restrict__ src, float4* __restrict__ dst, int n) {
  const int i = blockIdx.x * blockDim.x + threadIdx.x;
  if (i < n) dst[i] = src[i];
}

__global__ void scatter_kernel(const float* __restrict__ upd, const int* __restrict__ pidx,
                               const float* __restrict__ span_len, float* __restrict__ out_all) {
  const int k = blockIdx.x, b = blockIdx.y;
  if ((float)k < span_len[b]) {
    const int tgt = pidx[b * KK + k];
    const float* src = upd + ((size_t)b * KK + k) * DD;
    float* dst = out_all + ((size_t)b * NN + tgt) * DD;
    for (int d = threadIdx.x; d < DD; d += 256) dst[d] = src[d];
  }
}

// ---------------------------------------------------------------------------
extern "C" void kernel_launch(void* const* d_in, const int* in_sizes, int n_in,
                              void* d_out, int out_size, void* d_ws, size_t ws_size,
                              hipStream_t stream) {
  (void)in_sizes; (void)n_in; (void)out_size; (void)ws_size;

  const float* all_span   = (const float*)d_in[0];
  const float* update_in  = (const float*)d_in[1];
  const float* sq_mask    = (const float*)d_in[2];
  const float* span_len   = (const float*)d_in[3];
  const int*   span_begin = (const int*)d_in[4];
  const int*   prune_idx  = (const int*)d_in[6];
  const float* w_left  = (const float*)d_in[7];
  const float* b_left  = (const float*)d_in[8];
  const float* w_right = (const float*)d_in[9];
  const float* b_right = (const float*)d_in[10];
  const float* dist_e  = (const float*)d_in[11];
  const float* w_h     = (const float*)d_in[12];
  const float* b_h     = (const float*)d_in[13];
  const float* w_out   = (const float*)d_in[14];
  const float* b_out   = (const float*)d_in[15];
  const float* a_w     = (const float*)d_in[16];
  const float* gate_w  = (const float*)d_in[17];
  const float* gate_b  = (const float*)d_in[18];
  const float* pr_w    = (const float*)d_in[19];
  const float* pr_b    = (const float*)d_in[20];

  float* out_all = (float*)d_out;
  float* out_upd = out_all + (size_t)BB * NN * DD;
  float* out_rel = out_upd + (size_t)BB * KK * DD;

  char* wsp = (char*)d_ws;
  auto bump = [&](size_t bytes) -> void* {
    void* p = (void*)wsp;
    wsp += (bytes + 255) & ~(size_t)255;
    return p;
  };
  float*    left   = (float*)bump((size_t)BB * KK * HP * sizeof(float));
  float*    right  = (float*)bump((size_t)BB * KK * HP * sizeof(float));
  float*    ss     = (float*)bump((size_t)BB * KK * sizeof(float));
  float*    rel_ws = (float*)bump((size_t)BB * KK * KK * LL * sizeof(float));
  float*    updA   = (float*)bump((size_t)BB * KK * DD * sizeof(float));
  float*    updB   = (float*)bump((size_t)BB * KK * DD * sizeof(float));
  _Float16* whT    = (_Float16*)bump((size_t)HP * HP * sizeof(_Float16));
  _Float16* woT    = (_Float16*)bump((size_t)LP * HP * sizeof(_Float16));

  const size_t smem_pf =
      (size_t)(256 * HP + HP * HP + LP * HP + 8 * 32 * HP) * sizeof(_Float16) +
      (size_t)NBK * HP * sizeof(float);  // ~227 KB of the 320 KB WGP LDS

  prep_weights_kernel<<<(HP * HP + 255) / 256, 256, 0, stream>>>(w_h, w_out, whT, woT);
  proj_kernel<<<BB * KK, 256, 0, stream>>>(update_in, w_left, b_left, w_right, b_right,
                                           pr_w, pr_b, left, right, ss);
  pair_ff_kernel<<<dim3(KK / 16, KK / 16, BB), 256, smem_pf, stream>>>(
      left, right, ss, span_begin, dist_e, b_h, b_out, whT, woT, rel_ws);

  const float* ucur = update_in;
  float* bufs[2] = {updA, updB};
  for (int t = 0; t < 2; ++t) {
    float* unext = bufs[t];
    relprop_gate_kernel<<<dim3(KK, BB), 256, 0, stream>>>(
        rel_ws, sq_mask, ucur, a_w, gate_w, gate_b, span_len, unext);
    proj_kernel<<<BB * KK, 256, 0, stream>>>(unext, w_left, b_left, w_right, b_right,
                                             pr_w, pr_b, left, right, ss);
    float* rdst = (t == 1) ? out_rel : rel_ws;
    pair_ff_kernel<<<dim3(KK / 16, KK / 16, BB), 256, smem_pf, stream>>>(
        left, right, ss, span_begin, dist_e, b_h, b_out, whT, woT, rdst);
    ucur = unext;
  }

  copy4_kernel<<<(BB * NN * DD / 4 + 255) / 256, 256, 0, stream>>>(
      (const float4*)all_span, (float4*)out_all, BB * NN * DD / 4);
  scatter_kernel<<<dim3(KK, BB), 256, 0, stream>>>(ucur, prune_idx, span_len, out_all);
  hipMemcpyAsync(out_upd, ucur, (size_t)BB * KK * DD * sizeof(float),
                 hipMemcpyDeviceToDevice, stream);
}
